// CausalSelfAttention_26955214750354
// MI455X (gfx1250) — matrix operand order repaired and checked by reading, not compiled, not run on hardware
//
#include <hip/hip_runtime.h>

typedef __attribute__((ext_vector_type(16))) _Float16 v16h;
typedef __attribute__((ext_vector_type(8)))  _Float16 v8h;
typedef __attribute__((ext_vector_type(8)))  float    v8f;

#define BB   2
#define TT   2048
#define CC   1024
#define NH   16
#define HD   64
#define N3C  3072
#define WIN  256

#define KSTR 72   // padded LDS row stride (halves) for K tile: 144B = 9*16B
#define VSTR 40   // padded LDS row stride (halves) for V tile: 80B  = 5*16B

// ---------------------------------------------------------------------------
// CDNA5 async copy global->LDS (ASYNCcnt-tracked), 16 bytes per lane.
// ---------------------------------------------------------------------------
typedef __attribute__((address_space(3))) void lds_void;

static __device__ __forceinline__ uint32_t lds_off32(void* p) {
  return (uint32_t)(uintptr_t)(lds_void*)p;
}

static __device__ __forceinline__ void async_cp16(void* ldsdst, const void* gsrc) {
  uint32_t l = lds_off32(ldsdst);
  asm volatile("global_load_async_to_lds_b128 %0, %1, off"
               :: "v"(l), "v"(gsrc) : "memory");
}
static __device__ __forceinline__ void wait_async_le(int pending4) {
  if (pending4) asm volatile("s_wait_asynccnt 0x4" ::: "memory");
  else          asm volatile("s_wait_asynccnt 0x0" ::: "memory");
}

// ---------------------------------------------------------------------------
// WMMA helper: D = A(16x32 f16) * B(32x16 f16) + C(16x16 f32)
// ---------------------------------------------------------------------------
__device__ __forceinline__ v8f wmma_f16(v16h a, v16h b, v8f c) {
  return __builtin_amdgcn_wmma_f32_16x16x32_f16(
      false, a, false, b, (short)0, c, false, false);
}

// A fragment (16x32, MxK) from row-major [*, ld] f16 buffer.
__device__ __forceinline__ v16h frag_a_load(const _Float16* __restrict__ base,
                                            int ld, int row, int k0, int lane) {
  const _Float16* p = base + (size_t)row * ld + (k0 + ((lane >> 4) << 4));
  v8h lo = *(const v8h*)p;
  v8h hi = *(const v8h*)(p + 8);
  v16h r;
#pragma unroll
  for (int i = 0; i < 8; ++i) { r[i] = lo[i]; r[i + 8] = hi[i]; }
  return r;
}

// B fragment (32x16, KxN) from a K-contiguous (transposed [N, ldk]) buffer.
__device__ __forceinline__ v16h frag_b_loadT(const _Float16* __restrict__ baseT,
                                             int ldk, int col, int k0, int lane) {
  const _Float16* p = baseT + (size_t)col * ldk + (k0 + ((lane >> 4) << 4));
  return *(const v16h*)p;
}

// ---------------------------------------------------------------------------
// Stage 0: fp32 -> f16 conversion kernels
// ---------------------------------------------------------------------------
__global__ void k_cvt(const float* __restrict__ src, _Float16* __restrict__ dst, int n) {
  int i = blockIdx.x * blockDim.x + threadIdx.x;
  if (i < n) dst[i] = (_Float16)src[i];
}

__global__ void k_cvt_T(const float* __restrict__ src, _Float16* __restrict__ dst,
                        int K, int N) {
  int i = blockIdx.x * blockDim.x + threadIdx.x;
  if (i < K * N) {
    int n = i / K, k = i - n * K;
    dst[i] = (_Float16)src[(size_t)k * N + n];
  }
}

// ---------------------------------------------------------------------------
// Stage 1: QKV projection.  [4096,1024] x [1024,3072] ; 1 wave = 32x64 tile.
// ---------------------------------------------------------------------------
__global__ void __launch_bounds__(32)
k_qkv_gemm(const _Float16* __restrict__ xh, const _Float16* __restrict__ waT,
           const float* __restrict__ b_attn,
           _Float16* __restrict__ qh, _Float16* __restrict__ kh,
           _Float16* __restrict__ vT) {
  const int lane = threadIdx.x & 31;
  const int n0 = blockIdx.x * 64;
  const int m0 = blockIdx.y * 32;
  v8f acc[2][4];
#pragma unroll
  for (int i = 0; i < 2; ++i)
#pragma unroll
    for (int j = 0; j < 4; ++j) acc[i][j] = (v8f){};

  for (int k0 = 0; k0 < CC; k0 += 32) {
    v16h a0 = frag_a_load(xh, CC, m0 + (lane & 15), k0, lane);
    v16h a1 = frag_a_load(xh, CC, m0 + 16 + (lane & 15), k0, lane);
#pragma unroll
    for (int j = 0; j < 4; ++j) {
      v16h b = frag_b_loadT(waT, CC, n0 + j * 16 + (lane & 15), k0, lane);
      acc[0][j] = wmma_f16(a0, b, acc[0][j]);
      acc[1][j] = wmma_f16(a1, b, acc[1][j]);
    }
  }

  const int col   = lane & 15;
  const int rbase = (lane >> 4) << 3;
#pragma unroll
  for (int i = 0; i < 2; ++i) {
#pragma unroll
    for (int j = 0; j < 4; ++j) {
      const int n    = n0 + j * 16 + col;
      const float bn = b_attn[n];
      const int kind = n >> 10;
      const int cc   = n & 1023;
      const int h    = cc >> 6;
      const int d    = cc & 63;
#pragma unroll
      for (int r = 0; r < 8; ++r) {
        const int m = m0 + i * 16 + rbase + r;
        const int b = m >> 11;
        const int t = m & 2047;
        const float v = acc[i][j][r] + bn;
        const size_t hoff = (size_t)(b * NH + h);
        if (kind == 0)      qh[(hoff * TT + t) * HD + d] = (_Float16)(v * 0.125f);
        else if (kind == 1) kh[(hoff * TT + t) * HD + d] = (_Float16)v;
        else                vT[(hoff * HD + d) * TT + t] = (_Float16)v;
      }
    }
  }
}

// ---------------------------------------------------------------------------
// Stage 2: local-window flash attention.
// Block = 4 waves = 64 queries. K/V 32-key chunks staged into LDS with
// double-buffered async copies, shared by all 4 waves.
// ---------------------------------------------------------------------------
__global__ void __launch_bounds__(128)
k_attn(const _Float16* __restrict__ qh, const _Float16* __restrict__ kh,
       const _Float16* __restrict__ vT, _Float16* __restrict__ aoh) {
  __shared__ alignas(16) _Float16 sK[2][32 * KSTR];  // [key][d], padded
  __shared__ alignas(16) _Float16 sV[2][64 * VSTR];  // [d][key], padded
  __shared__ alignas(16) _Float16 sP[4][16 * 32];    // per-wave P transpose

  const int tid  = threadIdx.x;
  const int lane = tid & 31;
  const int wid  = tid >> 5;
  const int qb0  = blockIdx.x * 64;
  const int qt0  = qb0 + wid * 16;
  const int h    = blockIdx.y;
  const int b    = blockIdx.z;
  const size_t hoff = (size_t)(b * NH + h);
  const _Float16* Q  = qh + hoff * TT * HD;   // [T,D]
  const _Float16* Kg = kh + hoff * TT * HD;   // [T,D]
  const _Float16* Vg = vT + hoff * HD * TT;   // [D,T]

  const int col   = lane & 15;
  const int rbase = (lane >> 4) << 3;
  const int g16   = (lane >> 4) << 4;

  const v16h qa0 = frag_a_load(Q, HD, qt0 + col, 0,  lane);  // d 0..31
  const v16h qa1 = frag_a_load(Q, HD, qt0 + col, 32, lane);  // d 32..63

  v8f acc[4];
#pragma unroll
  for (int t = 0; t < 4; ++t) acc[t] = (v8f){};
  v8f mrow, lrow;
#pragma unroll
  for (int r = 0; r < 8; ++r) { mrow[r] = -1e30f; lrow[r] = 0.0f; }

  // Chunk copy: 4 async b128 per thread (2 for K tile, 2 for V tile).
  const int kkey = tid >> 3, kseg = tid & 7;      // K: 128 thr * 2 granules
  const int vd   = tid >> 1, vhf  = tid & 1;      // V: per-thread half row
  auto issue = [&](int s0, int buf) {
#pragma unroll
    for (int j = 0; j < 2; ++j) {
      const int key = kkey + j * 16;              // 0..31
      async_cp16(&sK[buf][key * KSTR + kseg * 8],
                 Kg + (size_t)(s0 + key) * HD + kseg * 8);
      async_cp16(&sV[buf][vd * VSTR + vhf * 16 + j * 8],
                 Vg + (size_t)vd * TT + s0 + vhf * 16 + j * 8);
    }
  };

  int s_first = qb0 - WIN; if (s_first < 0) s_first = 0; s_first &= ~31;
  const int s_last = qb0 + 32;   // last chunk start (qb0 % 64 == 0)

  issue(s_first, 0);
  int ib = 0;
  for (int s0 = s_first; s0 <= s_last; s0 += 32, ++ib) {
    const int buf = ib & 1;
    const bool more = (s0 + 32 <= s_last);
    if (more) issue(s0 + 32, buf ^ 1);
    wait_async_le(more ? 1 : 0);
    __syncthreads();   // chunk `buf` visible to all waves

    // Wave-uniform skip: does this chunk intersect this wave's window?
    if ((s0 <= qt0 + 15) && (s0 + 31 >= qt0 - WIN)) {
      // ---- scores: Q(16x64) x K_chunk^T(64x32) ----
      v8f sc[2];
#pragma unroll
      for (int hf = 0; hf < 2; ++hf) {
        const _Float16* kp = &sK[buf][(hf * 16 + col) * KSTR + g16];
        v16h kb0 = *(const v16h*)kp;          // d 0..31
        v16h kb1 = *(const v16h*)(kp + 32);   // d 32..63
        v8f s = (v8f){};
        s = wmma_f16(qa0, kb0, s);
        s = wmma_f16(qa1, kb1, s);
#pragma unroll
        for (int r = 0; r < 8; ++r) {
          const int qi = qt0 + rbase + r;
          const int kj = s0 + hf * 16 + col;
          const bool ok = (kj <= qi) && (kj >= qi - WIN);
          sc[hf][r] = ok ? s[r] : -1e30f;
        }
      }
      // ---- online softmax ----
      v8f mnew, corr;
#pragma unroll
      for (int r = 0; r < 8; ++r) {
        float v = fmaxf(sc[0][r], sc[1][r]);
        v = fmaxf(v, __shfl_xor(v, 1));
        v = fmaxf(v, __shfl_xor(v, 2));
        v = fmaxf(v, __shfl_xor(v, 4));
        v = fmaxf(v, __shfl_xor(v, 8));
        mnew[r] = fmaxf(mrow[r], v);
        corr[r] = __expf(mrow[r] - mnew[r]);
      }
#pragma unroll
      for (int hf = 0; hf < 2; ++hf)
#pragma unroll
        for (int r = 0; r < 8; ++r) sc[hf][r] = __expf(sc[hf][r] - mnew[r]);
#pragma unroll
      for (int r = 0; r < 8; ++r) {
        float v = sc[0][r] + sc[1][r];
        v += __shfl_xor(v, 1);
        v += __shfl_xor(v, 2);
        v += __shfl_xor(v, 4);
        v += __shfl_xor(v, 8);
        lrow[r] = lrow[r] * corr[r] + v;
        mrow[r] = mnew[r];
      }
#pragma unroll
      for (int t = 0; t < 4; ++t)
#pragma unroll
        for (int r = 0; r < 8; ++r) acc[t][r] *= corr[r];

      // ---- P: C-layout -> A-layout via per-wave LDS tile ----
      // LDS ops are in-order within a wave; wave_barrier stops reordering.
#pragma unroll
      for (int hf = 0; hf < 2; ++hf)
#pragma unroll
        for (int r = 0; r < 8; ++r)
          sP[wid][(rbase + r) * 32 + hf * 16 + col] = (_Float16)sc[hf][r];
      __builtin_amdgcn_wave_barrier();
      v16h pa;
      {
        const int pb = col * 32 + ((lane >> 4) << 4);
#pragma unroll
        for (int i = 0; i < 8; ++i) {
          pa[i]     = sP[wid][pb + i];
          pa[i + 8] = sP[wid][pb + 8 + i];
        }
      }
      __builtin_amdgcn_wave_barrier();

      // ---- PV: P(16x32) x V_chunk(32x64) ----
#pragma unroll
      for (int t = 0; t < 4; ++t) {
        v16h vb = *(const v16h*)&sV[buf][(t * 16 + col) * VSTR + g16];
        acc[t] = wmma_f16(pa, vb, acc[t]);
      }
    }
    __syncthreads();   // all waves done with `buf` before it is refilled
  }

  // ---- normalize + write f16 [B*T, C] for the projection GEMM ----
#pragma unroll
  for (int r = 0; r < 8; ++r) {
    const float rinv = 1.0f / lrow[r];
    const int ti = qt0 + rbase + r;
#pragma unroll
    for (int t = 0; t < 4; ++t) {
      const int c = h * HD + t * 16 + col;
      aoh[((size_t)b * TT + ti) * CC + c] = (_Float16)(acc[t][r] * rinv);
    }
  }
}

// ---------------------------------------------------------------------------
// Stage 3: output projection. [4096,1024] x [1024,1024] + bias -> fp32 out.
// ---------------------------------------------------------------------------
__global__ void __launch_bounds__(32)
k_proj_gemm(const _Float16* __restrict__ aoh, const _Float16* __restrict__ wpT,
            const float* __restrict__ b_proj, float* __restrict__ out) {
  const int lane = threadIdx.x & 31;
  const int n0 = blockIdx.x * 64;
  const int m0 = blockIdx.y * 32;
  v8f acc[2][4];
#pragma unroll
  for (int i = 0; i < 2; ++i)
#pragma unroll
    for (int j = 0; j < 4; ++j) acc[i][j] = (v8f){};

  for (int k0 = 0; k0 < CC; k0 += 32) {
    v16h a0 = frag_a_load(aoh, CC, m0 + (lane & 15), k0, lane);
    v16h a1 = frag_a_load(aoh, CC, m0 + 16 + (lane & 15), k0, lane);
#pragma unroll
    for (int j = 0; j < 4; ++j) {
      v16h b = frag_b_loadT(wpT, CC, n0 + j * 16 + (lane & 15), k0, lane);
      acc[0][j] = wmma_f16(a0, b, acc[0][j]);
      acc[1][j] = wmma_f16(a1, b, acc[1][j]);
    }
  }

  const int col   = lane & 15;
  const int rbase = (lane >> 4) << 3;
#pragma unroll
  for (int i = 0; i < 2; ++i)
#pragma unroll
    for (int j = 0; j < 4; ++j) {
      const int n = n0 + j * 16 + col;
      const float bn = b_proj[n];
#pragma unroll
      for (int r = 0; r < 8; ++r) {
        const int m = m0 + i * 16 + rbase + r;
        out[(size_t)m * CC + n] = acc[i][j][r] + bn;
      }
    }
}

// ---------------------------------------------------------------------------
extern "C" void kernel_launch(void* const* d_in, const int* in_sizes, int n_in,
                              void* d_out, int out_size, void* d_ws, size_t ws_size,
                              hipStream_t stream) {
  const float* x      = (const float*)d_in[0];  // [B,T,C]
  const float* W_attn = (const float*)d_in[1];  // [C,3C]
  const float* b_attn = (const float*)d_in[2];  // [3C]
  const float* W_proj = (const float*)d_in[3];  // [C,C]
  const float* b_proj = (const float*)d_in[4];  // [C]
  float* out = (float*)d_out;

  _Float16* ws = (_Float16*)d_ws;
  _Float16* xh  = ws;                                    // [B*T, C]
  _Float16* waT = xh  + (size_t)4096 * CC;               // [3C, C] = W_attn^T
  _Float16* wpT = waT + (size_t)N3C * CC;                // [C, C]  = W_proj^T
  _Float16* qh  = wpT + (size_t)CC * CC;                 // [B,H,T,D] (q * 1/8)
  _Float16* kh  = qh  + (size_t)BB * NH * TT * HD;       // [B,H,T,D]
  _Float16* vT  = kh  + (size_t)BB * NH * TT * HD;       // [B,H,D,T]
  _Float16* aoh = vT  + (size_t)BB * NH * TT * HD;       // [B*T, C]

  const int nx = 4096 * CC;
  k_cvt  <<<dim3((nx + 255) / 256),       256, 0, stream>>>(x, xh, nx);
  k_cvt_T<<<dim3((CC * N3C + 255) / 256), 256, 0, stream>>>(W_attn, waT, CC, N3C);
  k_cvt_T<<<dim3((CC * CC + 255) / 256),  256, 0, stream>>>(W_proj, wpT, CC, CC);

  k_qkv_gemm <<<dim3(N3C / 64, 4096 / 32), 32, 0, stream>>>(xh, waT, b_attn, qh, kh, vT);
  k_attn     <<<dim3(TT / 64, NH, BB),    128, 0, stream>>>(qh, kh, vT, aoh);
  k_proj_gemm<<<dim3(CC / 64, 4096 / 32),  32, 0, stream>>>(aoh, wpT, b_proj, out);
}
